// AttentionLayer_83545703842160
// MI455X (gfx1250) — matrix operand order repaired and checked by reading, not compiled, not run on hardware
//
#include <hip/hip_runtime.h>

#define BATCH 8
#define CH    64
#define NTOK  4096   // 64*64 spatial tokens

typedef __attribute__((ext_vector_type(16))) __bf16         v16bf;
typedef __attribute__((ext_vector_type(16))) unsigned short v16u;
typedef __attribute__((ext_vector_type(8)))  float          v8f;

static __device__ __forceinline__ unsigned short f2bf(float f) {
  union { float f; unsigned u; } c; c.f = f;
  unsigned r = c.u + 0x7FFFu + ((c.u >> 16) & 1u);   // round-to-nearest-even
  return (unsigned short)(r >> 16);
}

static __device__ __forceinline__ v16bf as_bf16(v16u u) {
  union { v16u u; v16bf b; } t; t.u = u; return t.b;
}

static __device__ __forceinline__ v16u lds_ld32B(const unsigned short* p) {
  union { uint4 q[2]; v16u v; } t;
  t.q[0] = *(const uint4*)p;
  t.q[1] = *(const uint4*)(p + 8);
  return t.v;
}

// ---------------------------------------------------------------------------
// gfx1250 async global->LDS copy (ASYNCcnt), 16B per lane, per-lane LDS dest.
// builtin -> inline asm -> synchronous fallback.
// Builtin signature (from hipcc diagnostic): first param is
// "int __vector(4) __device__ *", i.e. vector_size(16) int in addrspace(1).
// ---------------------------------------------------------------------------
#if defined(__HIP_DEVICE_COMPILE__) && __has_builtin(__builtin_amdgcn_global_load_async_to_lds_b128)
#define ASYNC_MODE 2
#elif defined(__HIP_DEVICE_COMPILE__) && defined(__gfx1250__)
#define ASYNC_MODE 1
#else
#define ASYNC_MODE 0
#endif

typedef int vi4 __attribute__((vector_size(16)));

static __device__ __forceinline__ void async_cp16(const void* g, void* l) {
#if ASYNC_MODE == 2
  __builtin_amdgcn_global_load_async_to_lds_b128(
      (__attribute__((address_space(1))) vi4*)g,
      (__attribute__((address_space(3))) vi4*)l, 0, 0);
#elif ASYNC_MODE == 1
  unsigned loff = (unsigned)(size_t)(__attribute__((address_space(3))) char*)l;
  asm volatile("global_load_async_to_lds_b128 %0, %1, off"
               :: "v"(loff), "v"(g) : "memory");
#else
  *(uint4*)l = *(const uint4*)g;
#endif
}

static __device__ __forceinline__ void wait_async() {
#if ASYNC_MODE == 2
#if __has_builtin(__builtin_amdgcn_s_wait_asynccnt)
  __builtin_amdgcn_s_wait_asynccnt(0);
#else
  asm volatile("s_wait_asynccnt 0x0" ::: "memory");
#endif
#elif ASYNC_MODE == 1
  asm volatile("s_wait_asynccnt 0x0" ::: "memory");
#endif
}

// ---------------------------------------------------------------------------
// Kernel 1: QKV projection.  One block = 64 tokens (one key-block) of one
// batch, 4 threads/token computing 16 channels each for q,k,v (fp32 math).
// q,k written [b][n][c] bf16; v written block-transposed [b][kb][c][tok] so
// the flash kernel can async-copy it linearly.
// ---------------------------------------------------------------------------
__global__ __launch_bounds__(256) void qkv_kernel(
    const float* __restrict__ x,
    const float* __restrict__ Wq, const float* __restrict__ bq,
    const float* __restrict__ Wk, const float* __restrict__ bk,
    const float* __restrict__ Wv, const float* __restrict__ bv,
    unsigned short* __restrict__ qo, unsigned short* __restrict__ ko,
    unsigned short* __restrict__ vto)
{
  __shared__ float Wl[3 * CH * CH];
  __shared__ float bl[3 * CH];
  __shared__ unsigned short vt_s[CH * 64];   // [dim][tok_local]

  for (int i = threadIdx.x; i < CH * CH; i += 256) {
    Wl[i]        = Wq[i];
    Wl[4096 + i] = Wk[i];
    Wl[8192 + i] = Wv[i];
  }
  if (threadIdx.x < CH) {
    bl[threadIdx.x]       = bq[threadIdx.x];
    bl[64 + threadIdx.x]  = bk[threadIdx.x];
    bl[128 + threadIdx.x] = bv[threadIdx.x];
  }
  __syncthreads();

  const int tl      = threadIdx.x >> 2;            // token local 0..63
  const int quarter = threadIdx.x & 3;             // 16 channels each
  const int token   = blockIdx.x * 64 + tl;
  const int b       = token >> 12;
  const int n       = token & (NTOK - 1);

  const float* xb = x + (size_t)b * CH * NTOK + n;
  float tv[CH];
#pragma unroll
  for (int kk = 0; kk < CH; ++kk) tv[kk] = xb[(size_t)kk * NTOK];

  unsigned qp[8], kp[8], vp[8];
  const int c0 = quarter * 16;
#pragma unroll
  for (int i = 0; i < 8; ++i) {
    unsigned qw = 0, kw = 0, vw = 0;
#pragma unroll
    for (int h = 0; h < 2; ++h) {
      int c = c0 + i * 2 + h;
      float aq = bl[c], ak = bl[64 + c], av = bl[128 + c];
#pragma unroll
      for (int kk = 0; kk < CH; ++kk) {
        float t = tv[kk];
        aq = fmaf(t, Wl[c * CH + kk], aq);
        ak = fmaf(t, Wl[4096 + c * CH + kk], ak);
        av = fmaf(t, Wl[8192 + c * CH + kk], av);
      }
      qw |= (unsigned)f2bf(aq) << (16 * h);
      kw |= (unsigned)f2bf(ak) << (16 * h);
      vw |= (unsigned)f2bf(av) << (16 * h);
    }
    qp[i] = qw; kp[i] = kw; vp[i] = vw;
  }

  size_t base = (size_t)token * CH + c0;           // 32B aligned
  *(uint4*)(qo + base)     = uint4{qp[0], qp[1], qp[2], qp[3]};
  *(uint4*)(qo + base + 8) = uint4{qp[4], qp[5], qp[6], qp[7]};
  *(uint4*)(ko + base)     = uint4{kp[0], kp[1], kp[2], kp[3]};
  *(uint4*)(ko + base + 8) = uint4{kp[4], kp[5], kp[6], kp[7]};

  // stage V transposed in LDS: vt_s[c][tok_local]
#pragma unroll
  for (int i = 0; i < 8; ++i) {
    vt_s[(c0 + 2 * i) * 64 + tl]     = (unsigned short)(vp[i] & 0xffffu);
    vt_s[(c0 + 2 * i + 1) * 64 + tl] = (unsigned short)(vp[i] >> 16);
  }
  __syncthreads();

  // linear coalesced writeback of the 8KB transposed block
  unsigned* dst = (unsigned*)(vto + (size_t)blockIdx.x * (CH * 64));
  const unsigned* src = (const unsigned*)vt_s;
  for (int i = threadIdx.x; i < 2048; i += 256) dst[i] = src[i];
}

// ---------------------------------------------------------------------------
// Kernel 2: fused flash attention, double-buffered async K/V staging.
// Block = 8 waves = 128 query rows of one batch; 64-key blocks.
// LDS map (53248 B):
//   [0,     9216)  K buf0 : 64 tok x (64+8 pad) dim bf16   (row = 144 B)
//   [9216, 18432)  K buf1
//   [18432,27648)  V^T buf0: 64 dim x (64+8 pad) tok bf16
//   [27648,36864)  V^T buf1
//   [36864,53248)  P tiles : 8 waves x 16 row x 64 key bf16
//   epilogue: O staging 128 x 66 f32 overlays [0,33792), barrier-guarded
// ---------------------------------------------------------------------------
__global__ __launch_bounds__(256) void flash_kernel(
    const unsigned short* __restrict__ qg,
    const unsigned short* __restrict__ kg,
    const unsigned short* __restrict__ vtg,
    float* __restrict__ out)
{
  __shared__ __align__(16) char smem[53248];
  unsigned short* Pb = (unsigned short*)(smem + 36864);

  const int tid  = threadIdx.x;
  const int wave = tid >> 5;
  const int lane = tid & 31;
  const int hl   = lane >> 4;      // lane half (K-group select)
  const int ln   = lane & 15;

  const int b     = blockIdx.x >> 5;
  const int q0    = (blockIdx.x & 31) * 128;
  const int qbase = q0 + wave * 16;

  unsigned short* Pw = Pb + wave * (16 * 64);

  // per-lane copy offsets: 512 16B-chunks per 8KB matrix, 64 per wave.
  // source chunk g -> row g>>3, chunk g&7 ; LDS row stride 144 B (pad 16 B).
  const int g0  = wave * 64 + lane;
  const int g1  = g0 + 32;
  const int l0  = (g0 >> 3) * 144 + (g0 & 7) * 16;
  const int l1  = (g1 >> 3) * 144 + (g1 & 7) * 16;
  const int go0 = g0 * 16;
  const int go1 = g1 * 16;

  const char* kbase = (const char*)(kg + (size_t)b * NTOK * CH);
  const char* vbase = (const char*)(vtg + (size_t)b * NTOK * CH);

  auto copy_block = [&](int buf, int kb) {
    char* kd = smem + (buf ? 9216 : 0);
    char* vd = smem + (buf ? 27648 : 18432);
    const char* ks = kbase + (size_t)kb * 8192;
    const char* vs = vbase + (size_t)kb * 8192;
    async_cp16(ks + go0, kd + l0);
    async_cp16(ks + go1, kd + l1);
    async_cp16(vs + go0, vd + l0);
    async_cp16(vs + go1, vd + l1);
  };

  // ---- load Q A-fragments (16x64, two 16x32 chunks), kept in registers ----
  v16u aq[2];
  const unsigned short* qrow = qg + ((size_t)b * NTOK + qbase + ln) * CH;
#pragma unroll
  for (int c = 0; c < 2; ++c) {
    union { unsigned d[8]; v16u u; } t;
#pragma unroll
    for (int vI = 0; vI < 8; ++vI) {
      int kk = c * 32 + hl * 16 + 2 * vI;
      t.d[vI] = *(const unsigned*)(qrow + kk);
    }
    aq[c] = t.u;
  }

  v8f zero = {0.f, 0.f, 0.f, 0.f, 0.f, 0.f, 0.f, 0.f};
  v8f o[4];
  float m[8], l[8];
#pragma unroll
  for (int dt = 0; dt < 4; ++dt) o[dt] = zero;
#pragma unroll
  for (int r = 0; r < 8; ++r) { m[r] = -3.0e38f; l[r] = 0.0f; }

  copy_block(0, 0);   // prologue: prefetch block 0 into buffer 0

  // ---------------------- key-block loop (64 keys/iter) --------------------
  for (int kb = 0; kb < NTOK / 64; ++kb) {
    const int cur = kb & 1;
    wait_async();      // my async chunks for buf[cur] have landed
    __syncthreads();   // everyone's landed; everyone done with buf[cur^1]

    if (kb + 1 < NTOK / 64) copy_block(cur ^ 1, kb + 1);  // prefetch next

    unsigned short* Kc = (unsigned short*)(smem + (cur ? 9216 : 0));
    unsigned short* Vc = (unsigned short*)(smem + (cur ? 27648 : 18432));

    // ---- S = (Q K^T) * 1/sqrt(C): four 16x16 tiles, K=64 contraction ----
    v8f s[4];
#pragma unroll
    for (int jt = 0; jt < 4; ++jt) {
      v8f acc = zero;
#pragma unroll
      for (int c = 0; c < 2; ++c) {
        v16u bkf = lds_ld32B(Kc + (jt * 16 + ln) * 72 + c * 32 + hl * 16);
        acc = __builtin_amdgcn_wmma_f32_16x16x32_bf16(
            false, as_bf16(aq[c]), false, as_bf16(bkf), (short)0, acc,
            false, false);
      }
      s[jt] = acc * 0.125f;   // 1/sqrt(64)
    }

    // ---- online softmax (rows spread across 16-lane halves) ----
    float alpha[8];
#pragma unroll
    for (int r = 0; r < 8; ++r) {
      float mx = fmaxf(fmaxf(s[0][r], s[1][r]), fmaxf(s[2][r], s[3][r]));
      mx = fmaxf(mx, __shfl_xor(mx, 1, 16));
      mx = fmaxf(mx, __shfl_xor(mx, 2, 16));
      mx = fmaxf(mx, __shfl_xor(mx, 4, 16));
      mx = fmaxf(mx, __shfl_xor(mx, 8, 16));
      float mn = fmaxf(m[r], mx);
      alpha[r] = __expf(m[r] - mn);
      m[r] = mn;
    }

    float rs[8];
#pragma unroll
    for (int r = 0; r < 8; ++r) rs[r] = 0.0f;
#pragma unroll
    for (int jt = 0; jt < 4; ++jt) {
#pragma unroll
      for (int r = 0; r < 8; ++r) {
        float p = __expf(s[jt][r] - m[r]);
        rs[r] += p;
        Pw[(r + 8 * hl) * 64 + jt * 16 + ln] = f2bf(p);
      }
    }
#pragma unroll
    for (int r = 0; r < 8; ++r) {
      float t = rs[r];
      t += __shfl_xor(t, 1, 16);
      t += __shfl_xor(t, 2, 16);
      t += __shfl_xor(t, 4, 16);
      t += __shfl_xor(t, 8, 16);
      l[r] = l[r] * alpha[r] + t;
    }
#pragma unroll
    for (int dt = 0; dt < 4; ++dt)
#pragma unroll
      for (int r = 0; r < 8; ++r) o[dt][r] *= alpha[r];

    // ---- O += P @ V : A-fragments of P (per-wave LDS), B from V^T ----
    v16u pa[2];
#pragma unroll
    for (int kc = 0; kc < 2; ++kc) {
      union { unsigned d[8]; v16u u; } ta;
#pragma unroll
      for (int vI = 0; vI < 8; ++vI) {
        int kk = kc * 32 + hl * 16 + 2 * vI;
        ta.d[vI] = *(const unsigned*)(Pw + ln * 64 + kk);
      }
      pa[kc] = ta.u;
    }
#pragma unroll
    for (int dt = 0; dt < 4; ++dt) {
      v8f acc = o[dt];
#pragma unroll
      for (int kc = 0; kc < 2; ++kc) {
        v16u bvf = lds_ld32B(Vc + (dt * 16 + ln) * 72 + kc * 32 + hl * 16);
        acc = __builtin_amdgcn_wmma_f32_16x16x32_bf16(
            false, as_bf16(pa[kc]), false, as_bf16(bvf), (short)0, acc,
            false, false);
      }
      o[dt] = acc;
    }
  }

  // ---- normalize, stage through LDS, coalesced [C][N] store ----
  float inv[8];
#pragma unroll
  for (int r = 0; r < 8; ++r) inv[r] = 1.0f / l[r];

  __syncthreads();                       // done with K/V/P regions
  float* Ost = (float*)smem;             // 128 x 66 f32
#pragma unroll
  for (int dt = 0; dt < 4; ++dt)
#pragma unroll
    for (int r = 0; r < 8; ++r)
      Ost[(wave * 16 + r + 8 * hl) * 66 + dt * 16 + ln] = o[dt][r] * inv[r];
  __syncthreads();

  float* ob = out + (size_t)b * CH * NTOK + q0;
  for (int i = tid; i < 128 * 64; i += 256) {
    int d = i >> 7, tl = i & 127;
    ob[(size_t)d * NTOK + tl] = Ost[tl * 66 + d];
  }
}

// ---------------------------------------------------------------------------
extern "C" void kernel_launch(void* const* d_in, const int* in_sizes, int n_in,
                              void* d_out, int out_size, void* d_ws, size_t ws_size,
                              hipStream_t stream) {
  (void)in_sizes; (void)n_in; (void)out_size; (void)ws_size;
  const float* x  = (const float*)d_in[0];
  const float* Wq = (const float*)d_in[1];
  const float* bq = (const float*)d_in[2];
  const float* Wk = (const float*)d_in[3];
  const float* bk = (const float*)d_in[4];
  const float* Wv = (const float*)d_in[5];
  const float* bv = (const float*)d_in[6];

  const size_t elems = (size_t)BATCH * NTOK * CH;   // 4 MiB each @ bf16
  unsigned short* qw  = (unsigned short*)d_ws;
  unsigned short* kw  = qw + elems;
  unsigned short* vtw = kw + elems;

  qkv_kernel<<<512, 256, 0, stream>>>(x, Wq, bq, Wk, bk, Wv, bv, qw, kw, vtw);
  flash_kernel<<<BATCH * (NTOK / 128), 256, 0, stream>>>(qw, kw, vtw,
                                                         (float*)d_out);
}
